// Update_57363583205517
// MI455X (gfx1250) — matrix operand order repaired and checked by reading, not compiled, not run on hardware
//
#include <hip/hip_runtime.h>

// ---------------------------------------------------------------------------
// Fused GRU-style node update for MI455X (gfx1250, wave32, WMMA).
//   m_proj = (X_m + X_cm) @ [W_rm;W_um;W_hm].T + b      (N x 384)
//   h_proj =  X_h         @ [W_rh;W_uh;W_hh].T + b      (N x 384)
//   r = sigmoid(rm+rh); u = sigmoid(um+uh); hhat = tanh(hm + r*hh)
//   out = (1-u)*X_h + u*hhat
// Memory-bound problem (~410MB HBM @ 23.3TB/s ~= 18us); bf16 WMMA with fp32
// accumulation is the only matrix path fast enough to stay under that floor.
// ---------------------------------------------------------------------------

typedef __attribute__((ext_vector_type(16))) __bf16 v16bf;
typedef __attribute__((ext_vector_type(4)))  __bf16 v4bf;
typedef __attribute__((ext_vector_type(8)))  float  v8f;
typedef __attribute__((ext_vector_type(4)))  float  v4f;

#define N_NODES   200000
#define D         128
#define LDS_STRIDE 136                 // 128 + 8 bf16 pad (272B rows -> bank-conflict-free ds_load_b128)
#define MAT_ELEMS (D * LDS_STRIDE)     // bf16 elems per staged weight matrix
#define LDS_BYTES (6 * MAT_ELEMS * 2)  // 208896 B of the 320KB/WGP LDS

__global__ __launch_bounds__(256) void gru_update_kernel(
    const float* __restrict__ Xm,  const float* __restrict__ Xcm, const float* __restrict__ Xh,
    const float* __restrict__ Wrm, const float* __restrict__ Wum, const float* __restrict__ Whm,
    const float* __restrict__ Wrh, const float* __restrict__ Wuh, const float* __restrict__ Whh,
    const float* __restrict__ brm, const float* __restrict__ bum, const float* __restrict__ bhm,
    const float* __restrict__ brh, const float* __restrict__ buh, const float* __restrict__ bhh,
    float* __restrict__ Out)
{
    extern __shared__ __bf16 lds[];    // gate order: rm, um, hm, rh, uh, hh

    const int tid = threadIdx.x;

    // ---- stage all six 128x128 fp32 weight matrices into LDS as bf16 ----
    {
        const float* const srcs[6] = { Wrm, Wum, Whm, Wrh, Wuh, Whh };
        #pragma unroll
        for (int g = 0; g < 6; ++g) {
            const float* W  = srcs[g];
            __bf16* dst     = lds + g * MAT_ELEMS;
            #pragma unroll 4
            for (int i = tid; i < D * (D / 4); i += 256) {  // 4096 float4 per matrix
                const int o  = i >> 5;                       // output row  (0..127)
                const int kq = i & 31;                       // float4 slot (0..31)
                v4f w = *(const v4f*)(W + o * D + kq * 4);
                v4bf b;
                b[0] = (__bf16)w.x; b[1] = (__bf16)w.y;
                b[2] = (__bf16)w.z; b[3] = (__bf16)w.w;
                *(v4bf*)(dst + o * LDS_STRIDE + kq * 4) = b;
            }
        }
    }
    __syncthreads();

    const int wave = tid >> 5;
    const int lane = tid & 31;
    const int half = lane >> 4;        // 0 for lanes 0-15, 1 for lanes 16-31
    const int mrow = lane & 15;

    const int rowBase = (blockIdx.x * 8 + wave) * 16;
    int arow = rowBase + mrow;
    if (arow >= N_NODES) arow = N_NODES - 1;   // clamp loads; stores guarded below

    const float* xm  = Xm  + (long)arow * D;
    const float* xcm = Xcm + (long)arow * D;
    const float* xhp = Xh  + (long)arow * D;

    // ---- build A fragments (16x32 bf16 per K-step) for both GEMM inputs ----
    // 16-bit A layout: lane(0-15)=M, K chunks [0,8)+[16,24); lane(16-31)=M, [8,16)+[24,32)
    v16bf aM[4], aH[4];
    #pragma unroll
    for (int kk = 0; kk < 4; ++kk) {
        const int c0 = kk * 32 + half * 8;
        const int c1 = c0 + 16;
        #pragma unroll
        for (int i = 0; i < 8; ++i) {
            aM[kk][i]     = (__bf16)(xm[c0 + i] + xcm[c0 + i]);
            aM[kk][8 + i] = (__bf16)(xm[c1 + i] + xcm[c1 + i]);
            aH[kk][i]     = (__bf16)xhp[c0 + i];
            aH[kk][8 + i] = (__bf16)xhp[c1 + i];
        }
    }

    // B layout: lane%16 = N column, lanes 0-15 hold K [0,8)+[16,24), lanes 16-31 hold K [8,16)+[24,32)
    const int bcol = lane & 15;
    const int bko  = half * 8;

    // ---- loop over the 8 column tiles of the d=128 output ----
    for (int j = 0; j < 8; ++j) {
        v8f acc[6];
        #pragma unroll
        for (int g = 0; g < 6; ++g)
            #pragma unroll
            for (int e = 0; e < 8; ++e) acc[g][e] = 0.0f;

        #pragma unroll
        for (int kk = 0; kk < 4; ++kk) {
            #pragma unroll
            for (int g = 0; g < 6; ++g) {
                union { uint4 u[2]; v16bf v; } b;
                const __bf16* p = lds + g * MAT_ELEMS
                                + (j * 16 + bcol) * LDS_STRIDE + kk * 32 + bko;
                b.u[0] = *(const uint4*)(p);
                b.u[1] = *(const uint4*)(p + 16);
                const v16bf a = (g < 3) ? aM[kk] : aH[kk];
                acc[g] = __builtin_amdgcn_wmma_f32_16x16x32_bf16(
                             false, a, false, b.v, (short)0, acc[g], false, false);
            }
        }

        // ---- fused epilogue for this 16x16 tile pair ----
        const int ncol = j * 16 + bcol;
        const float vbrm = brm[ncol], vbum = bum[ncol], vbhm = bhm[ncol];
        const float vbrh = brh[ncol], vbuh = buh[ncol], vbhh = bhh[ncol];

        #pragma unroll
        for (int r = 0; r < 8; ++r) {
            const int m   = half * 8 + r;     // C/D layout: lanes 0-15 rows 0-7, 16-31 rows 8-15
            const int row = rowBase + m;
            const float rv = acc[0][r] + vbrm + acc[3][r] + vbrh;
            const float uv = acc[1][r] + vbum + acc[4][r] + vbuh;
            const float hm = acc[2][r] + vbhm;
            const float hg = acc[5][r] + vbhh;
            const float reset = 1.0f / (1.0f + __expf(-rv));
            const float upd   = 1.0f / (1.0f + __expf(-uv));
            const float hhat  = tanhf(hm + reset * hg);
            if (row < N_NODES) {
                const float xhv = Xh[(long)row * D + ncol];   // exact fp32 blend input
                Out[(long)row * D + ncol] = (1.0f - upd) * xhv + upd * hhat;
            }
        }
    }
}

extern "C" void kernel_launch(void* const* d_in, const int* in_sizes, int n_in,
                              void* d_out, int out_size, void* d_ws, size_t ws_size,
                              hipStream_t stream) {
    const float* Xm  = (const float*)d_in[0];
    const float* Xcm = (const float*)d_in[1];
    const float* Xh  = (const float*)d_in[2];
    const float* Wrm = (const float*)d_in[3];   const float* brm = (const float*)d_in[4];
    const float* Wrh = (const float*)d_in[5];   const float* brh = (const float*)d_in[6];
    const float* Wum = (const float*)d_in[7];   const float* bum = (const float*)d_in[8];
    const float* Wuh = (const float*)d_in[9];   const float* buh = (const float*)d_in[10];
    const float* Whm = (const float*)d_in[11];  const float* bhm = (const float*)d_in[12];
    const float* Whh = (const float*)d_in[13];  const float* bhh = (const float*)d_in[14];
    float* Out = (float*)d_out;

    const int rowsPerBlock = 128;               // 8 wave32 x 16 rows
    const int grid = (N_NODES + rowsPerBlock - 1) / rowsPerBlock;   // 1563

    gru_update_kernel<<<grid, 256, LDS_BYTES, stream>>>(
        Xm, Xcm, Xh,
        Wrm, Wum, Whm, Wrh, Wuh, Whh,
        brm, bum, bhm, brh, buh, bhh,
        Out);
}
